// DigitCapsules_46926812676992
// MI455X (gfx1250) — matrix operand order repaired and checked by reading, not compiled, not run on hardware
//
#include <hip/hip_runtime.h>

// ---------------------------------------------------------------------------
// DigitCapsules dynamic routing, fused for MI455X (gfx1250, wave32, WMMA).
//
//   Pass A: W, x converted fp32 -> bf16 once into workspace (bandwidth-bound;
//           W_bf16 = 151 MB fits the 192 MB L2 for the routing passes).
//   Pass B: u_hat recomputed per routing iteration via v_wmma_f32_16x16x32_bf16
//           (never materialized: 604 MB fp32 would dominate HBM traffic).
//   3 routing iterations = 3 route passes + 3 reduce/squash passes.
//   No float atomics anywhere -> bitwise deterministic across replays.
// ---------------------------------------------------------------------------

typedef __attribute__((ext_vector_type(16))) __bf16          v16bf;
typedef __attribute__((ext_vector_type(16))) unsigned short  v16u;
typedef __attribute__((ext_vector_type(8)))  float           v8f;

#define B_    128
#define R_    1152
#define I_    64
#define CO_   1024     // C*O = 32*32
#define NR_   18       // r's per chunk
#define NCH_  64       // 1152 / 18

static __device__ __forceinline__ unsigned short f2bfbits(float f) {
  unsigned u = __builtin_bit_cast(unsigned, f);
  return (unsigned short)((u + 0x7FFFu + ((u >> 16) & 1u)) >> 16);  // RNE
}

// fp32 -> bf16 bulk conversion, 8 elements / thread (memory-bound).
__global__ __launch_bounds__(256) void f32_to_bf16(
    const float* __restrict__ in, unsigned short* __restrict__ out, int n8) {
  const int i = blockIdx.x * 256 + threadIdx.x;
  if (i >= n8) return;
  const float4* p = (const float4*)in + (size_t)i * 2;
  const float4 q0 = p[0], q1 = p[1];
  uint4 o;
  o.x = f2bfbits(q0.x) | ((unsigned)f2bfbits(q0.y) << 16);
  o.y = f2bfbits(q0.z) | ((unsigned)f2bfbits(q0.w) << 16);
  o.z = f2bfbits(q1.x) | ((unsigned)f2bfbits(q1.y) << 16);
  o.w = f2bfbits(q1.z) | ((unsigned)f2bfbits(q1.w) << 16);
  ((uint4*)out)[i] = o;
}

// Two 16-byte segments -> one 32-byte WMMA fragment (register pair).
static __device__ __forceinline__ v16u ld16(const unsigned short* p0,
                                            const unsigned short* p1) {
  struct P { uint4 a, b; } s;
  s.a = *(const uint4*)p0;
  s.b = *(const uint4*)p1;
  return __builtin_bit_cast(v16u, s);
}

// MODE 0: c uniform (1/32), no b update            -> s0 partials
// MODE 1: b_new = agreement(v0), write b, softmax  -> s1 partials
// MODE 2: b_new = b1 + agreement(v1), softmax      -> s2 partials
template <int MODE>
__global__ __launch_bounds__(256) void caps_route(
    const unsigned short* __restrict__ xb,  // [B, R, I]   bf16 bits
    const unsigned short* __restrict__ Wb,  // [R, C*O, I] bf16 bits
    const float* __restrict__ vprev,        // [B, C*O]
    float* __restrict__ blog,               // [B, R, C]
    float* __restrict__ spart)              // [NCH, 8, 16, 1024]
{
  __shared__ float bmat[16 * 32];   // b_new for this (b-tile, r)
  __shared__ float csm[16 * 32];    // softmax(b_new) over C

  const int tid  = threadIdx.x;
  const int lane = tid & 31;
  const int wave = tid >> 5;
  const int n    = lane & 15;   // WMMA N index / A-matrix M index
  const int hi   = lane >> 4;   // half-wave select
  const int b0   = blockIdx.x << 4;
  const int r0   = blockIdx.y * NR_;

  float sacc[8][8];             // thread-exclusive s accumulators
#pragma unroll
  for (int t = 0; t < 8; ++t)
#pragma unroll
    for (int v = 0; v < 8; ++v) sacc[t][v] = 0.f;

  float vv[8][8];               // hoisted v_prev values (L2-resident source)
  if (MODE >= 1) {
#pragma unroll
    for (int t = 0; t < 8; ++t) {
      const int co = (((wave << 3) + t) << 4) + n;
#pragma unroll
      for (int v = 0; v < 8; ++v)
        vv[t][v] = vprev[(size_t)(b0 + v + (hi << 3)) * CO_ + co];
    }
  }

  for (int r = r0; r < r0 + NR_; ++r) {
    // ---- A fragments: x_bf16[b0+n, r, :], ISA 16-bit A 16x32 layout ----
    const unsigned short* xr = xb + ((size_t)(b0 + n) * R_ + r) * I_ + hi * 8;
    const v16bf a0 = __builtin_bit_cast(v16bf, ld16(xr, xr + 16));
    const v16bf a1 = __builtin_bit_cast(v16bf, ld16(xr + 32, xr + 48));

    if (MODE >= 1) {
      __syncthreads();                    // prev-iteration consumers done
      for (int k = tid; k < 512; k += 256) {
        const int m = k >> 5, c = k & 31;
        bmat[k] = (MODE == 2) ? blog[((size_t)(b0 + m) * R_ + r) * 32 + c]
                              : 0.f;
      }
      __syncthreads();                    // bmat initialized
    }

    // ---- WMMA: U = X_r (16x64) * W_r^T (64x16) per co tile ----
    v8f acc[8];
    const unsigned short* wr0 = Wb + (size_t)r * CO_ * I_;
#pragma unroll
    for (int t = 0; t < 8; ++t) {
      const int co0 = (((wave << 3) + t) << 4);
      const unsigned short* wr = wr0 + (size_t)(co0 + n) * I_ + hi * 8;
      v8f cacc = {0.f, 0.f, 0.f, 0.f, 0.f, 0.f, 0.f, 0.f};
      cacc = __builtin_amdgcn_wmma_f32_16x16x32_bf16(
          false, a0, false, __builtin_bit_cast(v16bf, ld16(wr, wr + 16)),
          (short)0, cacc, false, false);
      cacc = __builtin_amdgcn_wmma_f32_16x16x32_bf16(
          false, a1, false, __builtin_bit_cast(v16bf, ld16(wr + 32, wr + 48)),
          (short)0, cacc, false, false);
      acc[t] = cacc;
    }

    if (MODE == 0) {
#pragma unroll
      for (int t = 0; t < 8; ++t)
#pragma unroll
        for (int v = 0; v < 8; ++v) sacc[t][v] += acc[t][v];
    } else {
      // ---- agreement: b_new[m,c] += sum_o u_hat[m,c,o] * v[m,c,o] ----
      // The two co-tiles of capsule c are combined before the cross-lane
      // reduce: 4 shuffles per (c, v) instead of 8.
#pragma unroll
      for (int q = 0; q < 4; ++q) {
        const int c = (wave << 2) + q;    // capsule owned by this wave
#pragma unroll
        for (int v = 0; v < 8; ++v) {
          float tv = acc[2 * q][v] * vv[2 * q][v] +
                     acc[2 * q + 1][v] * vv[2 * q + 1][v];
          tv += __shfl_xor(tv, 1, 32);     // reduce over 16-lane o/N dim,
          tv += __shfl_xor(tv, 2, 32);     // stays within half-wave (wave32)
          tv += __shfl_xor(tv, 4, 32);
          tv += __shfl_xor(tv, 8, 32);
          if (n == 0) bmat[((v + (hi << 3)) << 5) + c] += tv;
        }
      }
      __syncthreads();                    // agreement complete
      for (int k = tid; k < 512; k += 256) {  // persist b_new
        const int m = k >> 5, c = k & 31;
        blog[((size_t)(b0 + m) * R_ + r) * 32 + c] = bmat[k];
      }
      if (wave == 0 && lane < 16) {       // softmax over C=32 per batch row
        const int m = lane;
        float mx = -3.0e38f;
        for (int c = 0; c < 32; ++c) mx = fmaxf(mx, bmat[(m << 5) + c]);
        float sm = 0.f;
        for (int c = 0; c < 32; ++c) {
          const float e = __expf(bmat[(m << 5) + c] - mx);
          csm[(m << 5) + c] = e;
          sm += e;
        }
        const float inv = 1.f / sm;
        for (int c = 0; c < 32; ++c) csm[(m << 5) + c] *= inv;
      }
      __syncthreads();                    // csm ready
#pragma unroll
      for (int q = 0; q < 4; ++q) {
        const int c = (wave << 2) + q;
#pragma unroll
        for (int v = 0; v < 8; ++v) {
          const float cs = csm[((v + (hi << 3)) << 5) + c];  // one LDS read,
          sacc[2 * q][v]     += acc[2 * q][v] * cs;          // two FMAs
          sacc[2 * q + 1][v] += acc[2 * q + 1][v] * cs;
        }
      }
    }
  }

  // ---- write per-chunk s partials (thread-exclusive, no atomics) ----
  const float scale = (MODE == 0) ? (1.f / 32.f) : 1.f;
  float* outp = spart + ((size_t)(blockIdx.y * 8 + blockIdx.x) << 14);
#pragma unroll
  for (int t = 0; t < 8; ++t) {
    const int co0 = (((wave << 3) + t) << 4);
#pragma unroll
    for (int v = 0; v < 8; ++v)
      outp[(size_t)((v + (hi << 3)) << 10) + co0 + n] = sacc[t][v] * scale;
  }
}

// Sum the NCH_ chunk partials, then squash along O (lane == o).
__global__ __launch_bounds__(256) void caps_reduce_squash(
    const float* __restrict__ spart, float* __restrict__ out)
{
  const int lane = threadIdx.x & 31;
  const int w    = threadIdx.x >> 5;
  const int row  = blockIdx.x * 8 + w;        // row = b*32 + c, in [0,4096)
  const int b    = row >> 5;
  const int c    = row & 31;
  const int bt   = b >> 4, m = b & 15;
  const size_t off = (((size_t)bt * 16 + m) << 10) + (c << 5) + lane;

  float s = 0.f;
  for (int k = 0; k < NCH_; ++k) s += spart[((size_t)k << 17) + off];

  float n2 = s * s;                           // ||s||^2 over O via shuffles
  n2 += __shfl_xor(n2, 1, 32);
  n2 += __shfl_xor(n2, 2, 32);
  n2 += __shfl_xor(n2, 4, 32);
  n2 += __shfl_xor(n2, 8, 32);
  n2 += __shfl_xor(n2, 16, 32);

  const float norm = sqrtf(n2);
  const float f = n2 / ((1.f + n2) * (norm + 1e-8f));  // squash factor
  out[((size_t)row << 5) + lane] = s * f;
}

extern "C" void kernel_launch(void* const* d_in, const int* in_sizes, int n_in,
                              void* d_out, int out_size, void* d_ws,
                              size_t ws_size, hipStream_t stream) {
  (void)in_sizes; (void)n_in; (void)out_size; (void)ws_size;
  const float* x = (const float*)d_in[0];   // [128, 1152, 64]
  const float* W = (const float*)d_in[1];   // [1152, 32, 32, 64]
  float* out = (float*)d_out;               // [128, 32, 32]

  // ---- workspace layout (all 16-byte aligned) ----
  float* blog  = (float*)d_ws;                          //  4,718,592 f
  float* spart = blog + (size_t)B_ * R_ * 32;           //  8,388,608 f
  float* vbuf  = spart + (size_t)NCH_ * 8 * 16384;      //    131,072 f
  unsigned short* xbf = (unsigned short*)(vbuf + 131072);       //  9,437,184 h
  unsigned short* Wbf = xbf + (size_t)B_ * R_ * I_;             // 75,497,472 h

  const int nx8 = (B_ * R_ * I_) / 8;                   // 1,179,648
  const int nw8 = (R_ * CO_ * I_) / 8;                  // 9,437,184

  const dim3 grid(8, NCH_);
  const dim3 blk(256);

  // one-time fp32 -> bf16 conversion into workspace
  f32_to_bf16<<<nx8 / 256, 256, 0, stream>>>(x, xbf, nx8);
  f32_to_bf16<<<nw8 / 256, 256, 0, stream>>>(W, Wbf, nw8);

  // iter 0: uniform c -> s0 -> v0
  caps_route<0><<<grid, blk, 0, stream>>>(xbf, Wbf, vbuf, blog, spart);
  caps_reduce_squash<<<512, 256, 0, stream>>>(spart, vbuf);
  // iter 1: b1 = agr(v0); c1 = softmax(b1) -> s1 -> v1
  caps_route<1><<<grid, blk, 0, stream>>>(xbf, Wbf, vbuf, blog, spart);
  caps_reduce_squash<<<512, 256, 0, stream>>>(spart, vbuf);
  // iter 2: b2 = b1 + agr(v1); c2 = softmax(b2) -> s2 -> v2 (output)
  caps_route<2><<<grid, blk, 0, stream>>>(xbf, Wbf, vbuf, blog, spart);
  caps_reduce_squash<<<512, 256, 0, stream>>>(spart, out);
}
